// HANModel_87299505258755
// MI455X (gfx1250) — matrix operand order repaired and checked by reading, not compiled, not run on hardware
//
#include <hip/hip_runtime.h>
#include <hip/hip_bf16.h>

// ---------------------------------------------------------------------------
// HAN (hetero GAT) for MI455X / gfx1250, wave32 + WMMA bf16 path.
// Dense transforms: v_wmma_f32_16x16x32_bf16 (fp32 accumulate), B tiles staged
// in LDS via double-buffered global_load_async_to_lds_b128 (ASYNCcnt).
// Edge softmax/aggregation: fp32 global atomics over L2-resident buffers.
// ---------------------------------------------------------------------------

typedef __attribute__((ext_vector_type(16))) __bf16 v16bf;
typedef __attribute__((ext_vector_type(8)))  __bf16 v8bf;
typedef __attribute__((ext_vector_type(8)))  float  v8f;

#define ACT_NONE 0
#define ACT_RELU 1
#define ACT_TANH 2

#if defined(__has_builtin)
#if __has_builtin(__builtin_amdgcn_s_wait_asynccnt)
#define WAIT_ASYNC(n) __builtin_amdgcn_s_wait_asynccnt(n)
#endif
#endif
#ifndef WAIT_ASYNC
#define WAIT_ASYNC(n) asm volatile("s_wait_asynccnt %0" :: "i"(n) : "memory")
#endif

__device__ __forceinline__ float bf2f(unsigned short u) {
    return __uint_as_float(((unsigned int)u) << 16);
}
__device__ __forceinline__ unsigned short f2bf(float f) {
    unsigned int x = __float_as_uint(f);
    x += 0x7fffu + ((x >> 16) & 1u);   // round to nearest even
    return (unsigned short)(x >> 16);
}

// ---------------------------------------------------------------------------
// WMMA GEMM: C[M,N] = act(A[M,K](bf16) @ B[K,N](bf16) + bias[N])
// Block = 256 threads = 8 waves; wave w owns rows [blockIdx.y*128 + w*16, +16),
// cols [blockIdx.x*NTILES*16, +NTILES*16). K multiple of 32.
// NTILES=4 requires N%64==0; NTILES=1 requires N%16==0.
//
// Per 32-wide K step the block stages the 32 x (NTILES*16) B tile into LDS
// with async global->LDS copies (double buffered), then each wave reads its
// B fragments from LDS (lane&15 = N col, 16 K rows per lane; row stride
// padded +16B to spread LDS banks) and issues NTILES WMMAs, branch-free.
// A fragment (ISA 7.12.2, 16-bit A 16x32): lane&15 = M row; lanes 0-15 hold
// K = kk+0..7 / kk+16..23, lanes 16-31 hold kk+8..15 / kk+24..31.
// C/D: vgpr r -> M = row0 + r + (lane>=16 ? 8 : 0), N = col0 + (lane&15).
// ---------------------------------------------------------------------------
template <int NTILES>
__global__ __launch_bounds__(256)
void gemm_bf16_wmma(const unsigned short* __restrict__ A,
                    const unsigned short* __restrict__ B,
                    const float* __restrict__ bias,
                    float* __restrict__ Cf,
                    unsigned short* __restrict__ Cbf,
                    int M, int K, int N, int act)
{
    constexpr int TW     = NTILES * 16;        // cols per block
    constexpr int STRIDE = TW * 2 + 16;        // LDS row stride (bytes), padded
    constexpr int BUFSZ  = 32 * STRIDE;        // one K-step tile
    __shared__ __attribute__((aligned(32))) unsigned char ldsb[2 * BUFSZ];

    const int tid  = threadIdx.x;
    const int lane = tid & 31;
    const int wave = tid >> 5;
    const int row0 = blockIdx.y * 128 + wave * 16;
    const int col0 = blockIdx.x * TW;

    const int mA  = row0 + (lane & 15);
    const int mAc = (mA < M) ? mA : (M - 1);   // clamp tail rows; stores guarded
    const int kb  = (lane >> 4) << 3;          // 0 or 8

    // staging: 2*NTILES threads per row (16B chunks), 32 rows per K step
    const int      sthreads = 64 * NTILES;
    const int      srow     = tid / (2 * NTILES);
    const int      schunk   = tid % (2 * NTILES);
    const unsigned lds0     = (unsigned)(size_t)&ldsb[0];   // LDS byte offset

    auto stage = [&](int buf, int kk) {
        if (tid < sthreads) {
            const unsigned short* g = B + (size_t)(kk + srow) * N + col0 + schunk * 8;
            unsigned loff = lds0 + (unsigned)(buf * BUFSZ + srow * STRIDE + schunk * 16);
            asm volatile("global_load_async_to_lds_b128 %0, %1, off"
                         :: "v"(loff), "v"(g) : "memory");
        }
    };

    v8f acc[NTILES];
#pragma unroll
    for (int t = 0; t < NTILES; ++t) acc[t] = (v8f){0.f,0.f,0.f,0.f,0.f,0.f,0.f,0.f};

    const unsigned short* Arow = A + (size_t)mAc * K;
    const int nk = K / 32;
    stage(0, 0);

    for (int i = 0; i < nk; ++i) {
        const int kk  = i * 32;
        const int cur = i & 1;
        if (i + 1 < nk) {                       // prefetch next tile (double buffer)
            stage(cur ^ 1, kk + 32);
            WAIT_ASYNC(1);                      // current tile done; next in flight
        } else {
            WAIT_ASYNC(0);
        }
        __syncthreads();                        // tile visible to whole block

        union { v16bf v; v8bf h[2]; } af;
        af.h[0] = *(const v8bf*)(Arow + kk + kb);
        af.h[1] = *(const v8bf*)(Arow + kk + kb + 16);

        const unsigned char* bl = &ldsb[cur * BUFSZ + (lane & 15) * 2];
#pragma unroll
        for (int t = 0; t < NTILES; ++t) {
            v16bf bfrag;
#pragma unroll
            for (int s = 0; s < 16; ++s) {
                const int k = (s < 8) ? (kb + s) : (16 + kb + (s - 8));
                bfrag[s] = *(const __bf16*)(bl + k * STRIDE + t * 32);
            }
            acc[t] = __builtin_amdgcn_wmma_f32_16x16x32_bf16(
                         false, af.v, false, bfrag, (short)0, acc[t], false, false);
        }
        __syncthreads();                        // done reading before overwrite
    }

#pragma unroll
    for (int t = 0; t < NTILES; ++t) {
        const int n  = col0 + t * 16 + (lane & 15);
        const float bv = bias ? bias[n] : 0.0f;
#pragma unroll
        for (int r = 0; r < 8; ++r) {
            const int m = row0 + ((lane >> 4) << 3) + r;
            if (m >= M) continue;
            float v = acc[t][r] + bv;
            if (act == ACT_RELU)      v = fmaxf(v, 0.0f);
            else if (act == ACT_TANH) v = tanhf(v);
            const size_t o = (size_t)m * N + n;
            if (Cf)  Cf[o]  = v;
            if (Cbf) Cbf[o] = f2bf(v);
        }
    }
}

// ---------------------------------------------------------------------------
// Small utility / edge kernels
// ---------------------------------------------------------------------------
__global__ void fill_f32_kernel(float* p, float v, int n) {
    int i = blockIdx.x * blockDim.x + threadIdx.x;
    if (i < n) p[i] = v;
}
__global__ void fill_i32_kernel(int* p, int v, int n) {
    int i = blockIdx.x * blockDim.x + threadIdx.x;
    if (i < n) p[i] = v;
}
__global__ void cvt_bf16_kernel(const float* __restrict__ s, unsigned short* __restrict__ d, int n) {
    int i = blockIdx.x * blockDim.x + threadIdx.x;
    if (i < n) d[i] = f2bf(s[i]);
}
// pad columns Ns -> Nd with zeros (weights, row-major [K, Ns] -> [K, Nd] bf16)
__global__ void cvt2d_pad_bf16_kernel(const float* __restrict__ s, unsigned short* __restrict__ d,
                                      int K, int Ns, int Nd) {
    int i = blockIdx.x * blockDim.x + threadIdx.x;
    if (i >= K * Nd) return;
    int k = i / Nd, n = i % Nd;
    d[i] = (n < Ns) ? f2bf(s[(size_t)k * Ns + n]) : (unsigned short)0;
}
__global__ void pad_bias_kernel(const float* __restrict__ b, float* __restrict__ o, int Ns, int Nd) {
    int i = blockIdx.x * blockDim.x + threadIdx.x;
    if (i < Nd) o[i] = (i < Ns) ? b[i] : 0.0f;
}
// weff[k,h] = sum_c W[k, h*C+c] * a[h*C+c]   (pre-contract attention vector)
__global__ void weff_kernel(const float* __restrict__ W, const float* __restrict__ a,
                            float* __restrict__ weff, int K, int H, int C) {
    int i = blockIdx.x * blockDim.x + threadIdx.x;
    if (i >= K * H) return;
    int k = i / H, h = i % H;
    float s = 0.f;
    for (int c = 0; c < C; ++c) s += W[(size_t)k * (H * C) + h * C + c] * a[h * C + c];
    weff[i] = s;
}
// out[n,h] = sum_k Xbf[n,k] * weff[k,h]
__global__ void adot_kernel(const unsigned short* __restrict__ X, const float* __restrict__ weff,
                            float* __restrict__ o, int Nn, int K, int H) {
    int i = blockIdx.x * blockDim.x + threadIdx.x;
    if (i >= Nn * H) return;
    int n = i / H, h = i % H;
    const unsigned short* xr = X + (size_t)n * K;
    float s = 0.f;
    for (int k = 0; k < K; ++k) s += bf2f(xr[k]) * weff[k * H + h];
    o[i] = s;
}
// out[n,f] = b1[f] + b2[f]   (GAT biases are added post-aggregation)
__global__ void init_bias2_kernel(float* __restrict__ o, const float* __restrict__ b1,
                                  const float* __restrict__ b2, int Nn, int F) {
    int i = blockIdx.x * blockDim.x + threadIdx.x;
    if (i >= Nn * F) return;
    int f = i % F;
    o[i] = b1[f] + (b2 ? b2[f] : 0.0f);
}
// pass 1: leaky-relu logits + segment max (monotone int encoding for atomicMax)
__global__ void edge_logits_kernel(const int* __restrict__ src, const int* __restrict__ dst,
                                   const float* __restrict__ as_, const float* __restrict__ ad_,
                                   float* __restrict__ logit, int* __restrict__ mkey, int E, int H) {
    int i = blockIdx.x * blockDim.x + threadIdx.x;
    if (i >= E * H) return;
    int e = i / H, h = i % H;
    int s = src[e], d = dst[e];
    float l = as_[(size_t)s * H + h] + ad_[(size_t)d * H + h];
    l = (l > 0.f) ? l : 0.2f * l;
    logit[i] = l;
    int b = __float_as_int(l);
    int key = (b >= 0) ? b : (b ^ 0x7fffffff);
    atomicMax(&mkey[(size_t)d * H + h], key);
}
// pass 2: p = exp(l - m[dst]); segment sum
__global__ void edge_exp_kernel(const int* __restrict__ dst, float* __restrict__ pe,
                                const int* __restrict__ mkey, float* __restrict__ ssum, int E, int H) {
    int i = blockIdx.x * blockDim.x + threadIdx.x;
    if (i >= E * H) return;
    int e = i / H, h = i % H;
    int d = dst[e];
    int key = mkey[(size_t)d * H + h];
    int b = (key >= 0) ? key : (key ^ 0x7fffffff);
    float m = __int_as_float(b);
    float p = expf(pe[i] - m);
    pe[i] = p;
    atomicAdd(&ssum[(size_t)d * H + h], p);
}
// pass 3: out[dst, h*C+c] += (p/(s+1e-16)) * hs[src, h*C+c]
__global__ void edge_aggr_kernel(const int* __restrict__ src, const int* __restrict__ dst,
                                 const float* __restrict__ pe, const float* __restrict__ ssum,
                                 const unsigned short* __restrict__ hs, float* __restrict__ out,
                                 int E, int H, int C) {
    long long i = (long long)blockIdx.x * blockDim.x + threadIdx.x;
    long long total = (long long)E * H * C;
    if (i >= total) return;
    int c = (int)(i % C);
    long long t = i / C;
    int h = (int)(t % H);
    int e = (int)(t / H);
    int s = src[e], d = dst[e];
    float alpha = pe[(size_t)e * H + h] / (ssum[(size_t)d * H + h] + 1e-16f);
    float v = alpha * bf2f(hs[(size_t)s * (H * C) + h * C + c]);
    atomicAdd(&out[(size_t)d * (H * C) + h * C + c], v);
}
__global__ void elu_bf16_kernel(const float* __restrict__ s, unsigned short* __restrict__ d, int n) {
    int i = blockIdx.x * blockDim.x + threadIdx.x;
    if (i >= n) return;
    float x = s[i];
    d[i] = f2bf(x > 0.f ? x : (expf(x) - 1.0f));
}
// semantic attention: g = sigmoid(tanh(p2@W1+b1)@W2); pf = g*p2 (t precomputed)
__global__ void sem_kernel(const float* __restrict__ t, const float* __restrict__ W2,
                           const float* __restrict__ p2, unsigned short* __restrict__ pf, int Nn) {
    int n = blockIdx.x * blockDim.x + threadIdx.x;
    if (n >= Nn) return;
    const float* tr = t + (size_t)n * 128;
    float s = 0.f;
    for (int i = 0; i < 128; ++i) s += tr[i] * W2[i];
    float g = 1.0f / (1.0f + expf(-s));
    const float* pr = p2 + (size_t)n * 64;
    unsigned short* o = pf + (size_t)n * 64;
    for (int c = 0; c < 64; ++c) o[c] = f2bf(g * pr[c]);
}
__global__ void slice_kernel(const float* __restrict__ s, float* __restrict__ d,
                             int Nn, int Ns, int Nd) {
    int i = blockIdx.x * blockDim.x + threadIdx.x;
    if (i >= Nn * Nd) return;
    int n = i / Nd, j = i % Nd;
    d[i] = s[(size_t)n * Ns + j];
}

// ---------------------------------------------------------------------------
// Host orchestration
// ---------------------------------------------------------------------------
static inline int cdiv(int a, int b) { return (a + b - 1) / b; }

extern "C" void kernel_launch(void* const* d_in, const int* in_sizes, int n_in,
                              void* d_out, int out_size, void* d_ws, size_t ws_size,
                              hipStream_t stream)
{
    (void)in_sizes; (void)n_in; (void)out_size; (void)ws_size;
    const int NP = 50000, NA = 25000;
    const int EC = 250000, EW = 150000, EB = 150000;
    const int HID = 512;

    // --- input mapping: recursive dict-insertion order of setup_inputs() ---
    const float* xp   = (const float*)d_in[0];
    const float* xa   = (const float*)d_in[1];
    const int*   ec   = (const int*)d_in[2];    // [2,EC]: row0=src, row1=dst
    const int*   ew   = (const int*)d_in[3];
    const int*   eb   = (const int*)d_in[4];
    const float* Wpin = (const float*)d_in[5];
    const float* bpin = (const float*)d_in[6];
    const float* Wain = (const float*)d_in[7];
    const float* bain = (const float*)d_in[8];
    // conv1 base 9, per type {Ws,Wd,as,ad,b}, types: 0=cites,1=writes,2=wb
    auto c1 = [&](int t, int j) { return (const float*)d_in[9 + t * 5 + j]; };
    // conv2 base 24, per type {W,as,ad,b}, types: 0=cites,1=writes,2=wb
    auto c2 = [&](int t, int j) { return (const float*)d_in[24 + t * 4 + j]; };
    const float* W1sem = (const float*)d_in[36];
    const float* b1sem = (const float*)d_in[37];
    const float* W2sem = (const float*)d_in[38];
    const float* Wcls  = (const float*)d_in[42];
    const float* bcls  = (const float*)d_in[43];
    float* out = (float*)d_out;

    // --- workspace bump allocator ---
    char* base = (char*)d_ws; size_t off = 0;
    auto alloc = [&](size_t bytes) -> void* {
        void* p = base + off; off = (off + bytes + 255) & ~(size_t)255; return p;
    };
    unsigned short* xp_bf  = (unsigned short*)alloc((size_t)NP * 256 * 2);
    unsigned short* xa_bf  = (unsigned short*)alloc((size_t)NA * 128 * 2);
    unsigned short* hp_bf  = (unsigned short*)alloc((size_t)NP * 64 * 2);
    unsigned short* ha_bf  = (unsigned short*)alloc((size_t)NA * 64 * 2);
    unsigned short* Wpin_bf = (unsigned short*)alloc(256 * 64 * 2);
    unsigned short* Wain_bf = (unsigned short*)alloc(128 * 64 * 2);
    unsigned short* Ws_bf[3];
    for (int t = 0; t < 3; ++t) Ws_bf[t] = (unsigned short*)alloc(64 * 512 * 2);
    unsigned short* W2_bf[2];
    for (int t = 0; t < 2; ++t) W2_bf[t] = (unsigned short*)alloc(512 * 64 * 2);
    unsigned short* W1s_bf = (unsigned short*)alloc(64 * 128 * 2);
    unsigned short* Wcls_bf = (unsigned short*)alloc(64 * 16 * 2);
    float* bcls_pad = (float*)alloc(16 * 4);
    float* weffS = (float*)alloc(512 * 4);
    float* weffD = (float*)alloc(512 * 4);
    float* a_s   = (float*)alloc((size_t)NP * 8 * 4);
    float* a_d   = (float*)alloc((size_t)NP * 8 * 4);
    int*   mkey  = (int*)alloc((size_t)NP * 8 * 4);
    float* ssum  = (float*)alloc((size_t)NP * 8 * 4);
    float* ebuf  = (float*)alloc((size_t)EC * 8 * 4);
    unsigned short* hs_bf = (unsigned short*)alloc((size_t)NP * 512 * 2);
    float* p1acc = (float*)alloc((size_t)NP * 512 * 4);
    float* a1acc = (float*)alloc((size_t)NA * 512 * 4);
    unsigned short* p1_bf = (unsigned short*)alloc((size_t)NP * 512 * 2);
    unsigned short* a1_bf = (unsigned short*)alloc((size_t)NA * 512 * 2);
    float* p2acc = (float*)alloc((size_t)NP * 64 * 4);
    unsigned short* p2_bf = (unsigned short*)alloc((size_t)NP * 64 * 2);
    float* t_f32 = (float*)alloc((size_t)NP * 128 * 4);
    unsigned short* pf_bf = (unsigned short*)alloc((size_t)NP * 64 * 2);
    float* out_pad = (float*)alloc((size_t)NP * 16 * 4);

    const int B = 256;
    auto gemm = [&](const unsigned short* A, const unsigned short* Bm, const float* bias,
                    float* Cf, unsigned short* Cbf, int M, int K, int N, int act) {
        if (N % 64 == 0) {
            dim3 g(N / 64, cdiv(M, 128), 1);
            gemm_bf16_wmma<4><<<g, 256, 0, stream>>>(A, Bm, bias, Cf, Cbf, M, K, N, act);
        } else {
            dim3 g(cdiv(N, 16), cdiv(M, 128), 1);
            gemm_bf16_wmma<1><<<g, 256, 0, stream>>>(A, Bm, bias, Cf, Cbf, M, K, N, act);
        }
    };

    // --- phase 0: convert inputs + weights to bf16 ---
    cvt_bf16_kernel<<<cdiv(NP * 256, B), B, 0, stream>>>(xp, xp_bf, NP * 256);
    cvt_bf16_kernel<<<cdiv(NA * 128, B), B, 0, stream>>>(xa, xa_bf, NA * 128);
    cvt2d_pad_bf16_kernel<<<cdiv(256 * 64, B), B, 0, stream>>>(Wpin, Wpin_bf, 256, 64, 64);
    cvt2d_pad_bf16_kernel<<<cdiv(128 * 64, B), B, 0, stream>>>(Wain, Wain_bf, 128, 64, 64);
    for (int t = 0; t < 3; ++t)
        cvt2d_pad_bf16_kernel<<<cdiv(64 * 512, B), B, 0, stream>>>(c1(t, 0), Ws_bf[t], 64, 512, 512);
    for (int t = 0; t < 2; ++t)
        cvt2d_pad_bf16_kernel<<<cdiv(512 * 64, B), B, 0, stream>>>(c2(t, 0), W2_bf[t], 512, 64, 64);
    cvt2d_pad_bf16_kernel<<<cdiv(64 * 128, B), B, 0, stream>>>(W1sem, W1s_bf, 64, 128, 128);
    cvt2d_pad_bf16_kernel<<<cdiv(64 * 16, B), B, 0, stream>>>(Wcls, Wcls_bf, 64, 7, 16);
    pad_bias_kernel<<<1, 16, 0, stream>>>(bcls, bcls_pad, 7, 16);

    // --- phase 1: input projections (WMMA + fused bias/ReLU) ---
    gemm(xp_bf, Wpin_bf, bpin, nullptr, hp_bf, NP, 256, 64, ACT_RELU);
    gemm(xa_bf, Wain_bf, bain, nullptr, ha_bf, NA, 128, 64, ACT_RELU);

    // --- phase 2: conv1 (heads=8, ch=64). Accumulators pre-seeded with biases ---
    init_bias2_kernel<<<cdiv(NP * HID, B), B, 0, stream>>>(p1acc, c1(0, 4), c1(1, 4), NP, HID);
    init_bias2_kernel<<<cdiv(NA * HID, B), B, 0, stream>>>(a1acc, c1(2, 4), nullptr, NA, HID);

    auto run_gat1 = [&](const unsigned short* srcbf, int Ns, const unsigned short* dstbf, int Nd,
                        const int* e, int E, int t, float* outacc) {
        const int H = 8, C = 64;
        weff_kernel<<<cdiv(64 * H, B), B, 0, stream>>>(c1(t, 0), c1(t, 2), weffS, 64, H, C);
        weff_kernel<<<cdiv(64 * H, B), B, 0, stream>>>(c1(t, 1), c1(t, 3), weffD, 64, H, C);
        adot_kernel<<<cdiv(Ns * H, B), B, 0, stream>>>(srcbf, weffS, a_s, Ns, 64, H);
        adot_kernel<<<cdiv(Nd * H, B), B, 0, stream>>>(dstbf, weffD, a_d, Nd, 64, H);
        gemm(srcbf, Ws_bf[t], nullptr, nullptr, hs_bf, Ns, 64, HID, ACT_NONE);  // hs, no bias
        fill_i32_kernel<<<cdiv(Nd * H, B), B, 0, stream>>>(mkey, (int)0x80000000, Nd * H);
        fill_f32_kernel<<<cdiv(Nd * H, B), B, 0, stream>>>(ssum, 0.0f, Nd * H);
        const int* esrc = e; const int* edst = e + E;
        edge_logits_kernel<<<cdiv(E * H, B), B, 0, stream>>>(esrc, edst, a_s, a_d, ebuf, mkey, E, H);
        edge_exp_kernel<<<cdiv(E * H, B), B, 0, stream>>>(edst, ebuf, mkey, ssum, E, H);
        long long tot = (long long)E * H * C;
        edge_aggr_kernel<<<(int)((tot + B - 1) / B), B, 0, stream>>>(esrc, edst, ebuf, ssum,
                                                                     hs_bf, outacc, E, H, C);
    };
    run_gat1(hp_bf, NP, hp_bf, NP, ec, EC, 0, p1acc);   // cites: paper -> paper
    run_gat1(ha_bf, NA, hp_bf, NP, ew, EW, 1, p1acc);   // writes: author -> paper
    run_gat1(hp_bf, NP, ha_bf, NA, eb, EB, 2, a1acc);   // written_by: paper -> author

    // --- phase 3: ELU + bf16 for layer 2 ---
    elu_bf16_kernel<<<cdiv(NP * HID, B), B, 0, stream>>>(p1acc, p1_bf, NP * HID);
    elu_bf16_kernel<<<cdiv(NA * HID, B), B, 0, stream>>>(a1acc, a1_bf, NA * HID);

    // --- phase 4: conv2 (heads=1, ch=64, shared W for src/dst) ---
    init_bias2_kernel<<<cdiv(NP * 64, B), B, 0, stream>>>(p2acc, c2(0, 3), c2(1, 3), NP, 64);
    auto run_gat2 = [&](const unsigned short* srcbf, int Ns, const unsigned short* dstbf, int Nd,
                        const int* e, int E, int t) {
        const int H = 1, C = 64;
        weff_kernel<<<cdiv(HID * H, B), B, 0, stream>>>(c2(t, 0), c2(t, 1), weffS, HID, H, C);
        weff_kernel<<<cdiv(HID * H, B), B, 0, stream>>>(c2(t, 0), c2(t, 2), weffD, HID, H, C);
        adot_kernel<<<cdiv(Ns * H, B), B, 0, stream>>>(srcbf, weffS, a_s, Ns, HID, H);
        adot_kernel<<<cdiv(Nd * H, B), B, 0, stream>>>(dstbf, weffD, a_d, Nd, HID, H);
        gemm(srcbf, W2_bf[t], nullptr, nullptr, hs_bf, Ns, HID, 64, ACT_NONE);
        fill_i32_kernel<<<cdiv(Nd * H, B), B, 0, stream>>>(mkey, (int)0x80000000, Nd * H);
        fill_f32_kernel<<<cdiv(Nd * H, B), B, 0, stream>>>(ssum, 0.0f, Nd * H);
        const int* esrc = e; const int* edst = e + E;
        edge_logits_kernel<<<cdiv(E * H, B), B, 0, stream>>>(esrc, edst, a_s, a_d, ebuf, mkey, E, H);
        edge_exp_kernel<<<cdiv(E * H, B), B, 0, stream>>>(edst, ebuf, mkey, ssum, E, H);
        long long tot = (long long)E * H * C;
        edge_aggr_kernel<<<(int)((tot + B - 1) / B), B, 0, stream>>>(esrc, edst, ebuf, ssum,
                                                                     hs_bf, p2acc, E, H, C);
    };
    run_gat2(p1_bf, NP, p1_bf, NP, ec, EC, 0);          // cites
    run_gat2(a1_bf, NA, p1_bf, NP, ew, EW, 1);          // writes

    // --- phase 5: semantic attention + classifier (all GEMMs via WMMA) ---
    cvt_bf16_kernel<<<cdiv(NP * 64, B), B, 0, stream>>>(p2acc, p2_bf, NP * 64);
    gemm(p2_bf, W1s_bf, b1sem, t_f32, nullptr, NP, 64, 128, ACT_TANH);
    sem_kernel<<<cdiv(NP, B), B, 0, stream>>>(t_f32, W2sem, p2acc, pf_bf, NP);
    gemm(pf_bf, Wcls_bf, bcls_pad, out_pad, nullptr, NP, 64, 16, ACT_NONE);
    slice_kernel<<<cdiv(NP * 7, B), B, 0, stream>>>(out_pad, out, NP, 16, 7);
}
